// RelationalMemory_58274116272213
// MI455X (gfx1250) — matrix operand order repaired and checked by reading, not compiled, not run on hardware
//
#include <hip/hip_runtime.h>

#define B_   64
#define S_   2048
#define IN_  256
#define MS_  8
#define H_   8
#define HD_  64
#define M_   512
#define NG_  1024
#define TOPK_ 3
#define EPS_ 1e-5f

typedef unsigned short bfr;                                  // raw bf16 bits
typedef __attribute__((ext_vector_type(16))) __bf16 v16bf;   // WMMA A/B operand
typedef __attribute__((ext_vector_type(8)))  float  v8f;     // WMMA C/D operand

struct alignas(16) U4 { unsigned int x, y, z, w; };
struct U8 { U4 lo, hi; };
union V16U { v16bf v; U8 u; };

__device__ __forceinline__ float bf2f(unsigned int h) {
    unsigned int u = h << 16; float f; __builtin_memcpy(&f, &u, 4); return f;
}
__device__ __forceinline__ bfr f2bf(float f) {
    unsigned int u; __builtin_memcpy(&u, &f, 4);
    u += 0x7FFFu + ((u >> 16) & 1u);
    return (bfr)(u >> 16);
}

// ---------------------------------------------------------------------------
// Generic bf16 WMMA GEMM:  C[rows,N] = A[rows,K] @ Wt[N,K]^T + bias
// Wave owns a 16-row panel and loops over `ncpb` 64-column chunks so the A
// panel is re-read from L1/WGP$ (not HBM). Per K-step: stage A + 4 B tiles
// into distinct registers, then 4 back-to-back WMMAs (loads overlap compute).
// ---------------------------------------------------------------------------
__global__ __launch_bounds__(256)
void wmma_gemm_kernel(const bfr* __restrict__ A, int lda,
                      const bfr* __restrict__ Wt, int ldw,
                      const float* __restrict__ bias,
                      float* __restrict__ Cf, bfr* __restrict__ Cb, int ldc,
                      int rows, int K, int relu, int ncpb)
{
    const int lane = threadIdx.x & 31;
    const int wave = threadIdx.x >> 5;
    const int m0   = blockIdx.x * 128 + wave * 16;
    if (m0 >= rows) return;                       // wave-uniform
    const int lm = lane & 15;
    const int lh = lane >> 4;

    int arow = m0 + lm; if (arow >= rows) arow = rows - 1;
    const bfr* ap = A + (size_t)arow * lda + 16 * lh;

    for (int nc = 0; nc < ncpb; ++nc) {
        const int n0 = (blockIdx.y * ncpb + nc) * 64;
        const bfr* bp0 = Wt + (size_t)(n0 + lm) * ldw + 16 * lh;
        const bfr* bp1 = bp0 + (size_t)16 * ldw;
        const bfr* bp2 = bp0 + (size_t)32 * ldw;
        const bfr* bp3 = bp0 + (size_t)48 * ldw;

        v8f zero = {0.f,0.f,0.f,0.f,0.f,0.f,0.f,0.f};
        v8f acc[4]; acc[0]=zero; acc[1]=zero; acc[2]=zero; acc[3]=zero;

#pragma unroll 2
        for (int k0 = 0; k0 < K; k0 += 32) {
            V16U a, b[4];
            a.u.lo    = *(const U4*)(ap + k0);
            a.u.hi    = *(const U4*)(ap + k0 + 8);
            b[0].u.lo = *(const U4*)(bp0 + k0);
            b[0].u.hi = *(const U4*)(bp0 + k0 + 8);
            b[1].u.lo = *(const U4*)(bp1 + k0);
            b[1].u.hi = *(const U4*)(bp1 + k0 + 8);
            b[2].u.lo = *(const U4*)(bp2 + k0);
            b[2].u.hi = *(const U4*)(bp2 + k0 + 8);
            b[3].u.lo = *(const U4*)(bp3 + k0);
            b[3].u.hi = *(const U4*)(bp3 + k0 + 8);
#pragma unroll
            for (int t = 0; t < 4; ++t)
                acc[t] = __builtin_amdgcn_wmma_f32_16x16x32_bf16(
                    false, a.v, false, b[t].v, (short)0, acc[t], false, false);
        }
#pragma unroll
        for (int t = 0; t < 4; ++t) {
            int n = n0 + t * 16 + lm;
            float bb = bias ? bias[n] : 0.f;
#pragma unroll
            for (int r = 0; r < 8; ++r) {
                int mrow = m0 + r + 8 * lh;
                if (mrow < rows) {
                    float v = acc[t][r] + bb;
                    if (relu) v = fmaxf(v, 0.f);
                    if (Cf) Cf[(size_t)mrow * ldc + n] = v;
                    if (Cb) Cb[(size_t)mrow * ldc + n] = f2bf(v);
                }
            }
        }
    }
}

// --------------------------- small helper kernels ---------------------------

__global__ void conv_f32_bf16_kernel(const float* __restrict__ s,
                                     bfr* __restrict__ d, long long n) {
    long long i = (long long)blockIdx.x * 256 + threadIdx.x;
    if (i < n) d[i] = f2bf(s[i]);
}

// memory -> bf16 [B*MS,512]   and  tanh(memory) -> bf16 [MS,B,512] (slot-major)
__global__ void memconv_kernel(const float* __restrict__ mem,
                               bfr* __restrict__ memb, bfr* __restrict__ tmem) {
    int i = blockIdx.x * 256 + threadIdx.x;
    if (i < B_ * MS_ * M_) {
        float v = mem[i];
        memb[i] = f2bf(v);
        int j = i & (M_ - 1);
        int bm = i >> 9;                 // M_ = 512
        int b = bm / MS_, m = bm % MS_;
        tmem[((size_t)m * B_ + b) * M_ + j] = f2bf(tanhf(v));
    }
}

// W [K,N] f32 -> Wt [N,K] bf16
__global__ void wtrans_kernel(const float* __restrict__ s, bfr* __restrict__ d,
                              int K, int N) {
    long long i = (long long)blockIdx.x * 256 + threadIdx.x;
    if (i < (long long)K * N) {
        int k = (int)(i / N), n = (int)(i % N);
        d[(size_t)n * K + k] = f2bf(s[i]);
    }
}

// partial sums of relu(rw * x) over 128-step chunks -> gpart[16,B,512]
__global__ __launch_bounds__(256)
void gate_acc_kernel(const bfr* __restrict__ x, const float* __restrict__ rw,
                     float* __restrict__ gpart) {
    int blk = blockIdx.x;
    int b = blk >> 4, sc = blk & 15;
    for (int j = threadIdx.x; j < M_; j += 256) {
        float w = rw[j];
        float sum = 0.f;
        const bfr* xp = x + ((size_t)b * S_ + (size_t)sc * 128) * M_ + j;
        for (int s = 0; s < 128; ++s)
            sum += fmaxf(w * bf2f(xp[(size_t)s * M_]), 0.f);
        gpart[((size_t)sc * B_ + b) * M_ + j] = sum;
    }
}

__global__ void act_reduce_kernel(const float* __restrict__ gpart,
                                  bfr* __restrict__ actb) {
    int i = blockIdx.x * 256 + threadIdx.x;
    if (i < B_ * M_) {
        int b = i / M_, j = i % M_;
        float s = 0.f;
        for (int sc = 0; sc < 16; ++sc) s += gpart[((size_t)sc * B_ + b) * M_ + j];
        actb[i] = f2bf(s * (1.f / S_));
    }
}

// ------------------- attention: logits + softmax + top-3 --------------------
// grid = B*H blocks; wave w handles memory slot w. Emits weighted-x rows and
// probability mass (exact softmax via online max/sum, top-3 unrenormalized).
__global__ __launch_bounds__(256)
void attn_topk_kernel(const float* __restrict__ qf,   // [B*MS,512]
                      const bfr* __restrict__ kmat,   // [B*S,512] bf16
                      const bfr* __restrict__ x,      // [B*S,512] bf16
                      bfr* __restrict__ xw,           // [H,B*MS,512] bf16
                      float* __restrict__ mass)       // [B,H,MS]
{
    const int bh = blockIdx.x;
    const int b = bh / H_, h = bh % H_;
    const int lane = threadIdx.x & 31;
    const int wave = threadIdx.x >> 5;   // slot

    __shared__ float qs[MS_][HD_];
    __shared__ float rmax[MS_][32];
    __shared__ float rsum[MS_][32];
    __shared__ float rval[MS_][32][TOPK_];
    __shared__ int   ridx[MS_][32][TOPK_];
    __shared__ float psh[MS_][TOPK_];
    __shared__ int   ssh[MS_][TOPK_];

    for (int i = threadIdx.x; i < MS_ * HD_; i += 256) {
        int m = i >> 6, d = i & 63;
        qs[m][d] = qf[(size_t)(b * MS_ + m) * M_ + h * HD_ + d];
    }
    __syncthreads();

    const int m = wave;
    float lmax = -3.0e38f, lsum = 0.f;
    float t0 = -3.0e38f, t1 = -3.0e38f, t2 = -3.0e38f;
    int   i0 = 0, i1 = 0, i2 = 0;
    for (int s = lane; s < S_; s += 32) {
        const U4* kp = (const U4*)(kmat + (size_t)(b * S_ + s) * M_ + h * HD_);
        float acc = 0.f;
#pragma unroll
        for (int c = 0; c < 8; ++c) {
            U4 kk = kp[c];
            const float* qp = &qs[m][c * 8];
            acc += qp[0]*bf2f(kk.x & 0xffffu) + qp[1]*bf2f(kk.x >> 16)
                 + qp[2]*bf2f(kk.y & 0xffffu) + qp[3]*bf2f(kk.y >> 16)
                 + qp[4]*bf2f(kk.z & 0xffffu) + qp[5]*bf2f(kk.z >> 16)
                 + qp[6]*bf2f(kk.w & 0xffffu) + qp[7]*bf2f(kk.w >> 16);
        }
        if (acc > lmax) { lsum = lsum * __expf(lmax - acc) + 1.f; lmax = acc; }
        else            { lsum += __expf(acc - lmax); }
        if (acc > t0)      { t2=t1;i2=i1; t1=t0;i1=i0; t0=acc;i0=s; }
        else if (acc > t1) { t2=t1;i2=i1; t1=acc;i1=s; }
        else if (acc > t2) { t2=acc;i2=s; }
    }
    rmax[m][lane] = lmax; rsum[m][lane] = lsum;
    rval[m][lane][0]=t0; rval[m][lane][1]=t1; rval[m][lane][2]=t2;
    ridx[m][lane][0]=i0; ridx[m][lane][1]=i1; ridx[m][lane][2]=i2;
    __syncthreads();

    if (threadIdx.x < MS_) {
        int mm = threadIdx.x;
        float gmax = -3.0e38f;
        for (int l = 0; l < 32; ++l) gmax = fmaxf(gmax, rmax[mm][l]);
        float gsum = 0.f;
        for (int l = 0; l < 32; ++l) gsum += rsum[mm][l] * __expf(rmax[mm][l] - gmax);
        float v0=-3.0e38f, v1=-3.0e38f, v2=-3.0e38f; int j0=0, j1=0, j2=0;
        for (int l = 0; l < 32; ++l)
            for (int i = 0; i < TOPK_; ++i) {
                float v = rval[mm][l][i]; int id = ridx[mm][l][i];
                if (v > v0)      { v2=v1;j2=j1; v1=v0;j1=j0; v0=v;j0=id; }
                else if (v > v1) { v2=v1;j2=j1; v1=v;j1=id; }
                else if (v > v2) { v2=v;j2=id; }
            }
        float inv = 1.f / gsum;
        float p0 = __expf(v0 - gmax) * inv;
        float p1 = __expf(v1 - gmax) * inv;
        float p2 = __expf(v2 - gmax) * inv;
        psh[mm][0]=p0; psh[mm][1]=p1; psh[mm][2]=p2;
        ssh[mm][0]=j0; ssh[mm][1]=j1; ssh[mm][2]=j2;
        mass[((size_t)b * H_ + h) * MS_ + mm] = p0 + p1 + p2;
    }
    __syncthreads();

    float p0 = psh[m][0], p1 = psh[m][1], p2 = psh[m][2];
    const bfr* xr0 = x + (size_t)(b * S_ + ssh[m][0]) * M_;
    const bfr* xr1 = x + (size_t)(b * S_ + ssh[m][1]) * M_;
    const bfr* xr2 = x + (size_t)(b * S_ + ssh[m][2]) * M_;
    bfr* xwr = xw + ((size_t)h * (B_ * MS_) + b * MS_ + m) * M_;
    for (int j = lane; j < M_; j += 32) {
        float v = p0 * bf2f(xr0[j]) + p1 * bf2f(xr1[j]) + p2 * bf2f(xr2[j]);
        xwr[j] = f2bf(v);
    }
}

// mem1 = LN(memory + att + mass*bv; g1,b1)  -> f32 + bf16
__global__ __launch_bounds__(256)
void ln1_kernel(const float* __restrict__ mem, const float* __restrict__ att,
                const float* __restrict__ mass, const float* __restrict__ bv,
                const float* __restrict__ gam, const float* __restrict__ bet,
                float* __restrict__ mem1f, bfr* __restrict__ mem1b)
{
    const int r = blockIdx.x;
    const int b = r / MS_, m = r % MS_;
    __shared__ float t[M_];
    __shared__ float sb[256];
    float ls = 0.f, lq = 0.f;
    for (int j = threadIdx.x; j < M_; j += 256) {
        int h = j >> 6;
        float v = mem[(size_t)r * M_ + j] + att[(size_t)r * M_ + j]
                + mass[((size_t)b * H_ + h) * MS_ + m] * bv[j];
        t[j] = v; ls += v; lq += v * v;
    }
    sb[threadIdx.x] = ls; __syncthreads();
    for (int s = 128; s > 0; s >>= 1) { if (threadIdx.x < s) sb[threadIdx.x] += sb[threadIdx.x + s]; __syncthreads(); }
    float mean = sb[0] / M_; __syncthreads();
    sb[threadIdx.x] = lq; __syncthreads();
    for (int s = 128; s > 0; s >>= 1) { if (threadIdx.x < s) sb[threadIdx.x] += sb[threadIdx.x + s]; __syncthreads(); }
    float var = sb[0] / M_ - mean * mean;
    float rstd = rsqrtf(var + EPS_);
    for (int j = threadIdx.x; j < M_; j += 256) {
        float v = (t[j] - mean) * rstd * gam[j] + bet[j];
        mem1f[(size_t)r * M_ + j] = v;
        mem1b[(size_t)r * M_ + j] = f2bf(v);
    }
}

// LN(mem1 + mlp2; g2,b2) -> tanh -> gated combine -> out
__global__ __launch_bounds__(256)
void final_kernel(const float* __restrict__ mem1f, const float* __restrict__ mlp2,
                  const float* __restrict__ gam, const float* __restrict__ bet,
                  const float* __restrict__ gmem,   // [MS,B,1024]
                  const float* __restrict__ gin,    // [B,1024]
                  const float* __restrict__ memory,
                  const float* __restrict__ fbias, const float* __restrict__ ibias,
                  float* __restrict__ out)
{
    const int r = blockIdx.x;
    const int b = r / MS_, m = r % MS_;
    __shared__ float t[M_];
    __shared__ float sb[256];
    float ls = 0.f, lq = 0.f;
    for (int j = threadIdx.x; j < M_; j += 256) {
        float v = mem1f[(size_t)r * M_ + j] + mlp2[(size_t)r * M_ + j];
        t[j] = v; ls += v; lq += v * v;
    }
    sb[threadIdx.x] = ls; __syncthreads();
    for (int s = 128; s > 0; s >>= 1) { if (threadIdx.x < s) sb[threadIdx.x] += sb[threadIdx.x + s]; __syncthreads(); }
    float mean = sb[0] / M_; __syncthreads();
    sb[threadIdx.x] = lq; __syncthreads();
    for (int s = 128; s > 0; s >>= 1) { if (threadIdx.x < s) sb[threadIdx.x] += sb[threadIdx.x + s]; __syncthreads(); }
    float var = sb[0] / M_ - mean * mean;
    float rstd = rsqrtf(var + EPS_);
    float fb = fbias[0], ib = ibias[0];
    const float* gm = gmem + ((size_t)m * B_ + b) * NG_;
    const float* gi = gin + (size_t)b * NG_;
    for (int j = threadIdx.x; j < M_; j += 256) {
        float nm = (t[j] - mean) * rstd * gam[j] + bet[j];
        float igate = 1.f / (1.f + __expf(-(gm[j] + gi[j] + ib)));
        float fgate = 1.f / (1.f + __expf(-(gm[M_ + j] + gi[M_ + j] + fb)));
        out[(size_t)r * M_ + j] = igate * tanhf(nm) + fgate * memory[(size_t)r * M_ + j];
    }
}

// ---------------------------------------------------------------------------
extern "C" void kernel_launch(void* const* d_in, const int* in_sizes, int n_in,
                              void* d_out, int out_size, void* d_ws, size_t ws_size,
                              hipStream_t stream) {
    (void)in_sizes; (void)n_in; (void)out_size; (void)ws_size;
    const float* inputs = (const float*)d_in[0];
    const float* memory = (const float*)d_in[1];
    const float* Wq = (const float*)d_in[2];  const float* bq = (const float*)d_in[3];
    const float* Wk = (const float*)d_in[4];  const float* bk = (const float*)d_in[5];
    const float* Wv = (const float*)d_in[6];  const float* bv = (const float*)d_in[7];
    const float* Wm = (const float*)d_in[8];  const float* bm = (const float*)d_in[9];
    const float* pg1 = (const float*)d_in[10]; const float* pb1 = (const float*)d_in[11];
    const float* pg2 = (const float*)d_in[12]; const float* pb2 = (const float*)d_in[13];
    const float* Wi = (const float*)d_in[14]; const float* bi = (const float*)d_in[15];
    const float* rw = (const float*)d_in[16];
    const float* Wr = (const float*)d_in[17]; const float* br = (const float*)d_in[18];
    const float* Wg = (const float*)d_in[19];
    const float* fbias = (const float*)d_in[20];
    const float* ibias = (const float*)d_in[21];
    float* out = (float*)d_out;

    char* wp = (char*)d_ws;
    auto carve = [&](size_t bytes) -> void* {
        void* r = (void*)wp;
        wp += (bytes + 255) & ~((size_t)255);
        return r;
    };
    bfr*   xb    = (bfr*)carve((size_t)B_ * S_ * IN_ * 2);     // inputs bf16
    bfr*   xproj = (bfr*)carve((size_t)B_ * S_ * M_ * 2);      // x bf16
    bfr*   kb    = (bfr*)carve((size_t)B_ * S_ * M_ * 2);      // k bf16
    bfr*   WiT   = (bfr*)carve((size_t)M_ * IN_ * 2);
    bfr*   WkT   = (bfr*)carve((size_t)M_ * M_ * 2);
    bfr*   WqT   = (bfr*)carve((size_t)M_ * M_ * 2);
    bfr*   WvT   = (bfr*)carve((size_t)M_ * M_ * 2);
    bfr*   WmT   = (bfr*)carve((size_t)M_ * M_ * 2);
    bfr*   WrT   = (bfr*)carve((size_t)NG_ * M_ * 2);
    bfr*   WgT   = (bfr*)carve((size_t)MS_ * NG_ * M_ * 2);
    bfr*   memb  = (bfr*)carve((size_t)B_ * MS_ * M_ * 2);
    bfr*   tmem  = (bfr*)carve((size_t)MS_ * B_ * M_ * 2);
    float* qf    = (float*)carve((size_t)B_ * MS_ * M_ * 4);
    bfr*   xw    = (bfr*)carve((size_t)H_ * B_ * MS_ * M_ * 2);
    float* massf = (float*)carve((size_t)B_ * H_ * MS_ * 4);
    float* attf  = (float*)carve((size_t)B_ * MS_ * M_ * 4);
    float* mem1f = (float*)carve((size_t)B_ * MS_ * M_ * 4);
    bfr*   mem1b = (bfr*)carve((size_t)B_ * MS_ * M_ * 2);
    bfr*   mlp1b = (bfr*)carve((size_t)B_ * MS_ * M_ * 2);
    float* mlp2f = (float*)carve((size_t)B_ * MS_ * M_ * 4);
    float* gpart = (float*)carve((size_t)16 * B_ * M_ * 4);
    bfr*   actb  = (bfr*)carve((size_t)B_ * M_ * 2);
    float* gin   = (float*)carve((size_t)B_ * NG_ * 4);
    float* gmem  = (float*)carve((size_t)MS_ * B_ * NG_ * 4);

    auto gemm = [&](const bfr* A, int lda, const bfr* Wt, int ldw, const float* bias,
                    float* Cf, bfr* Cb, int ldc, int rowsv, int Nv, int Kv,
                    int reluv, int ncpb) {
        dim3 g((rowsv + 127) / 128, Nv / 64 / ncpb);
        wmma_gemm_kernel<<<g, 256, 0, stream>>>(A, lda, Wt, ldw, bias, Cf, Cb, ldc,
                                                rowsv, Kv, reluv, ncpb);
    };

    // ---- precision prep ----
    long long nin = (long long)B_ * S_ * IN_;
    conv_f32_bf16_kernel<<<(unsigned)((nin + 255) / 256), 256, 0, stream>>>(inputs, xb, nin);
    memconv_kernel<<<(B_ * MS_ * M_ + 255) / 256, 256, 0, stream>>>(memory, memb, tmem);
    wtrans_kernel<<<(unsigned)(((long long)IN_ * M_ + 255) / 256), 256, 0, stream>>>(Wi, WiT, IN_, M_);
    wtrans_kernel<<<(unsigned)(((long long)M_ * M_ + 255) / 256), 256, 0, stream>>>(Wk, WkT, M_, M_);
    wtrans_kernel<<<(unsigned)(((long long)M_ * M_ + 255) / 256), 256, 0, stream>>>(Wq, WqT, M_, M_);
    wtrans_kernel<<<(unsigned)(((long long)M_ * M_ + 255) / 256), 256, 0, stream>>>(Wv, WvT, M_, M_);
    wtrans_kernel<<<(unsigned)(((long long)M_ * M_ + 255) / 256), 256, 0, stream>>>(Wm, WmT, M_, M_);
    wtrans_kernel<<<(unsigned)(((long long)M_ * NG_ + 255) / 256), 256, 0, stream>>>(Wr, WrT, M_, NG_);
    for (int m = 0; m < MS_; ++m)
        wtrans_kernel<<<(unsigned)(((long long)M_ * NG_ + 255) / 256), 256, 0, stream>>>(
            Wg + (size_t)m * M_ * NG_, WgT + (size_t)m * NG_ * M_, M_, NG_);

    // ---- big WMMA GEMMs (ncpb=8: A panel re-read from L1, HBM traffic 1x) ----
    gemm(xb, IN_, WiT, IN_, bi, nullptr, xproj, M_, B_ * S_, M_, IN_, 0, 8);  // x
    gemm(xproj, M_, WkT, M_, bk, nullptr, kb, M_, B_ * S_, M_, M_, 0, 8);     // k
    gemm(memb, M_, WqT, M_, bq, qf, nullptr, M_, B_ * MS_, M_, M_, 0, 1);     // q

    // ---- gate_in activations (deterministic two-stage reduction) ----
    gate_acc_kernel<<<B_ * 16, 256, 0, stream>>>(xproj, rw, gpart);
    act_reduce_kernel<<<(B_ * M_ + 255) / 256, 256, 0, stream>>>(gpart, actb);

    // ---- attention: softmax + top-3 -> weighted x rows ----
    attn_topk_kernel<<<B_ * H_, 256, 0, stream>>>(qf, kb, xproj, xw, massf);

    // att = xw @ Wv_head  (per-head WMMA; mass*bv folded into LN1)
    for (int h = 0; h < H_; ++h)
        gemm(xw + (size_t)h * B_ * MS_ * M_, M_, WvT + (size_t)h * HD_ * M_, M_,
             nullptr, attf + h * HD_, nullptr, M_, B_ * MS_, HD_, M_, 0, 1);

    // ---- memory update path ----
    ln1_kernel<<<B_ * MS_, 256, 0, stream>>>(memory, attf, massf, bv, pg1, pb1, mem1f, mem1b);
    gemm(mem1b, M_, WmT, M_, bm, nullptr, mlp1b, M_, B_ * MS_, M_, M_, 1, 1);  // mlp1 (relu)
    gemm(mlp1b, M_, WmT, M_, bm, mlp2f, nullptr, M_, B_ * MS_, M_, M_, 1, 1);  // mlp2 (relu)

    // ---- gates ----
    gemm(actb, M_, WrT, M_, br, gin, nullptr, NG_, B_, NG_, M_, 0, 1);         // gate_in
    for (int m = 0; m < MS_; ++m)                                              // gate_mem
        gemm(tmem + (size_t)m * B_ * M_, M_, WgT + (size_t)m * NG_ * M_, M_,
             nullptr, gmem + (size_t)m * B_ * NG_, nullptr, NG_, B_, NG_, M_, 0, 1);

    // ---- final LN + gated combine ----
    final_kernel<<<B_ * MS_, 256, 0, stream>>>(mem1f, mlp2f, pg2, pb2, gmem, gin,
                                               memory, fbias, ibias, out);
}
